// BoundaryPointTransformer_Unet_PrimSeg_48095043780757
// MI455X (gfx1250) — matrix operand order repaired and checked by reading, not compiled, not run on hardware
//
#include <hip/hip_runtime.h>

typedef __attribute__((ext_vector_type(16))) _Float16 v16h;
typedef __attribute__((ext_vector_type(4)))  _Float16 h4;
typedef __attribute__((ext_vector_type(8)))  float    v8f;

#define CCH 64
#define NSB 16

static __device__ __forceinline__ v8f wmma_f16(v16h a, v16h b, v8f c) {
  // D = A(16x32 f16) * B(32x16 f16) + C(16x16 f32)
  return __builtin_amdgcn_wmma_f32_16x16x32_f16(false, a, false, b, (short)0, c, false, false);
}

// ---------------------------------------------------------------------------
// Kernel A: xq/xk/xv = x @ W{q,k,v}.T + b  via v_wmma_f32_16x16x32_f16
// One wave per 16-row tile; 3 matrices x 4 n-tiles x 2 k-chunks = 24 WMMAs.
// x tile staged via GLOBAL_LOAD_ASYNC_TO_LDS_B128 (ASYNCcnt), weights as f16.
// ---------------------------------------------------------------------------
__global__ void __launch_bounds__(256) proj_qkv_wmma(
    const float* __restrict__ x,
    const float* __restrict__ Wq, const float* __restrict__ bq,
    const float* __restrict__ Wk, const float* __restrict__ bk,
    const float* __restrict__ Wv, const float* __restrict__ bv,
    float* __restrict__ xq, float* __restrict__ xk, float* __restrict__ xv,
    int npts)
{
  __shared__ __align__(16) _Float16 sW[3 * CCH * CCH];  // f16 weights W[c][k]
  __shared__ __align__(16) float    sXf[8][16 * CCH];   // per-wave x tile, f32

  const int tid  = threadIdx.x;
  const int w    = tid >> 5, lane = tid & 31;
  const int tile = blockIdx.x * 8 + w;
  const int row0 = tile * 16;
  const bool anyrow   = row0 < npts;
  const bool fulltile = (row0 + 16) <= npts;

  // ---- async-stage the 16x64 f32 x tile into this wave's LDS region ----
  if (fulltile) {
    unsigned lbase = (unsigned)(uintptr_t)&sXf[w][0];
    const float* gbase = x + (size_t)row0 * CCH;        // 256B-aligned rows
    #pragma unroll
    for (int it = 0; it < 8; ++it) {                    // 8 x 512B per wave
      unsigned off = (unsigned)(it * 512 + lane * 16);
      unsigned la  = lbase + off;
      asm volatile("global_load_async_to_lds_b128 %0, %1, %2"
                   :: "v"(la), "v"(off), "s"(gbase)
                   : "memory");
    }
  } else if (anyrow) {                                  // partial-tile fallback
    for (int i = lane; i < 16 * CCH; i += 32) {
      int r = i >> 6, c = i & 63;
      int gr = row0 + r; if (gr >= npts) gr = npts - 1;
      sXf[w][i] = x[(size_t)gr * CCH + c];
    }
  }

  // ---- stage f16 weights (vectorized b128 loads, b64 LDS stores) ----
  for (int i4 = tid; i4 < (CCH * CCH) / 4; i4 += 256) {
    float4 q4 = ((const float4*)Wq)[i4];
    float4 k4 = ((const float4*)Wk)[i4];
    float4 v4 = ((const float4*)Wv)[i4];
    h4 hq = { (_Float16)q4.x, (_Float16)q4.y, (_Float16)q4.z, (_Float16)q4.w };
    h4 hk = { (_Float16)k4.x, (_Float16)k4.y, (_Float16)k4.z, (_Float16)k4.w };
    h4 hv = { (_Float16)v4.x, (_Float16)v4.y, (_Float16)v4.z, (_Float16)v4.w };
    *(h4*)&sW[i4 * 4]                 = hq;
    *(h4*)&sW[CCH * CCH + i4 * 4]     = hk;
    *(h4*)&sW[2 * CCH * CCH + i4 * 4] = hv;
  }
  __syncthreads();
  if (!anyrow) return;                 // whole-wave uniform; no barriers after
  asm volatile("s_wait_asynccnt 0" ::: "memory");   // x tile landed in LDS

  const int row = lane & 15, hi = lane >> 4, col = row;

  // A fragments (16x32 f16 layout: lanes 0-15 K=0..15, lanes 16-31 K=16..31)
  v16h a0, a1;
  #pragma unroll
  for (int e = 0; e < 16; ++e) {
    int kl = 16 * hi + e;
    a0[e] = (_Float16)sXf[w][row * CCH + kl];
    a1[e] = (_Float16)sXf[w][row * CCH + 32 + kl];
  }

  const int kbb = 16 * hi;  // B 32x16: lanes0-15 K=0..15, lanes16-31 K=16..31
  #pragma unroll
  for (int m = 0; m < 3; ++m) {
    const _Float16* wb  = sW + m * CCH * CCH;
    const float* bias   = (m == 0) ? bq : (m == 1 ? bk : bv);
    float*       op     = (m == 0) ? xq : (m == 1 ? xk : xv);
    #pragma unroll
    for (int nt = 0; nt < 4; ++nt) {
      v16h b0, b1;
      #pragma unroll
      for (int e = 0; e < 16; ++e) {
        b0[e] = wb[(nt * 16 + col) * CCH + kbb + e];        // B[k][n] = W[n][k]
        b1[e] = wb[(nt * 16 + col) * CCH + 32 + kbb + e];
      }
      v8f acc = {};
      acc = wmma_f16(a0, b0, acc);
      acc = wmma_f16(a1, b1, acc);
      float bb = bias[nt * 16 + col];
      #pragma unroll
      for (int i = 0; i < 8; ++i) {     // D: (M=i+8*hi, N=col)
        int gr = row0 + i + 8 * hi;
        if (gr < npts) op[(size_t)gr * CCH + nt * 16 + col] = acc[i] + bb;
      }
    }
  }
}

// ---------------------------------------------------------------------------
// Kernel B: fused gather + positional MLP + attention MLP (WMMA, neighbors
// as the M dimension) + softmax + grouped weighted reduce. One wave / point.
// ---------------------------------------------------------------------------
__global__ void __launch_bounds__(256) attn_fused_wmma(
    const float* __restrict__ p, const long long* __restrict__ idx,
    const float* __restrict__ xq, const float* __restrict__ xk,
    const float* __restrict__ xv,
    const float* __restrict__ p_w1, const float* __restrict__ p_b1,
    const float* __restrict__ p_bn_g, const float* __restrict__ p_bn_b,
    const float* __restrict__ p_bn_m, const float* __restrict__ p_bn_v,
    const float* __restrict__ p_w2, const float* __restrict__ p_b2,
    const float* __restrict__ w_bn1_g, const float* __restrict__ w_bn1_b,
    const float* __restrict__ w_bn1_m, const float* __restrict__ w_bn1_v,
    const float* __restrict__ w_w1, const float* __restrict__ w_b1,
    const float* __restrict__ w_bn2_g, const float* __restrict__ w_bn2_b,
    const float* __restrict__ w_bn2_m, const float* __restrict__ w_bn2_v,
    const float* __restrict__ w_w2, const float* __restrict__ w_b2,
    float* __restrict__ out, int npts)
{
  __shared__ __align__(16) _Float16 sw1h[8 * CCH];      // w_w1 f16, [h][k]
  __shared__ _Float16 sw2h[64];                         // w_w2 f16, [h][k]
  __shared__ float sbn1s[64], sbn1b[64];                // folded BN1
  __shared__ float sbn2s[8],  sbn2b[8];                 // folded BN2
  __shared__ float sp2[192], spb2[64];                  // p_w2 (C,3), p_b2
  __shared__ float sph[8][48];                          // 16 nbrs x 3 (pos MLP)
  __shared__ int   sidx[8][16];
  __shared__ __align__(16) _Float16 swpre[8][NSB * CCH]; // BN1+ReLU(w), GEMM1 A
  __shared__ __align__(16) float    svbuf[8][NSB * CCH]; // xv + p_r
  __shared__ float sw1buf[8][128];                      // GEMM1 result 16x8
  __shared__ float swn[8][128];                         // softmax weights 16x8

  const int tid = threadIdx.x;
  for (int i = tid; i < 8 * CCH; i += 256) sw1h[i] = (_Float16)w_w1[i];
  if (tid < 64) sw2h[tid] = (_Float16)w_w2[tid];
  if (tid < 64) {
    float sc = w_bn1_g[tid] * rsqrtf(w_bn1_v[tid] + 1e-5f);
    sbn1s[tid] = sc;
    sbn1b[tid] = w_bn1_b[tid] - w_bn1_m[tid] * sc;
    spb2[tid]  = p_b2[tid];
  }
  if (tid < 8) {
    float sc = w_bn2_g[tid] * rsqrtf(w_bn2_v[tid] + 1e-5f);
    sbn2s[tid] = sc;
    sbn2b[tid] = w_bn2_b[tid] - w_bn2_m[tid] * sc;
  }
  if (tid < 192) sp2[tid] = p_w2[tid];
  __syncthreads();

  const int w = tid >> 5, lane = tid & 31;
  int n = blockIdx.x * 8 + w;
  if (n >= npts) n = npts - 1;   // keep barriers uniform (N%8==0 anyway)

  // ---- stage 1: neighbor ids + 3-dim positional MLP ----
  if (lane < 16) {
    long long nb = idx[(size_t)n * NSB + lane];
    int nbi = (int)nb;
    sidx[w][lane] = nbi;
    float d0 = p[(size_t)nbi * 3 + 0] - p[(size_t)n * 3 + 0];
    float d1 = p[(size_t)nbi * 3 + 1] - p[(size_t)n * 3 + 1];
    float d2 = p[(size_t)nbi * 3 + 2] - p[(size_t)n * 3 + 2];
    #pragma unroll
    for (int t = 0; t < 3; ++t) {
      float h  = p_b1[t] + p_w1[t * 3] * d0 + p_w1[t * 3 + 1] * d1 + p_w1[t * 3 + 2] * d2;
      float sc = p_bn_g[t] * rsqrtf(p_bn_v[t] + 1e-5f);
      float hb = (h - p_bn_m[t]) * sc + p_bn_b[t];
      sph[w][lane * 3 + t] = fmaxf(hb, 0.0f);
    }
    __builtin_prefetch(xk + (size_t)nbi * CCH, 0, 0);  // global_prefetch_b8
    __builtin_prefetch(xv + (size_t)nbi * CCH, 0, 0);
  }
  __syncthreads();

  // ---- stage 2: L2-resident gather (b128), build w_pre (f16) and v (f32) ----
  {
    const float4* xqr = (const float4*)(xq + (size_t)n * CCH);
    for (int f4 = lane; f4 < (NSB * CCH) / 4; f4 += 32) {  // 256 float4 chunks
      int j = f4 >> 4, c4 = f4 & 15, c = c4 * 4;
      int nb = sidx[w][j];
      float4 kk = ((const float4*)(xk + (size_t)nb * CCH))[c4];
      float4 vv = ((const float4*)(xv + (size_t)nb * CCH))[c4];
      float4 qq = xqr[c4];
      float h0 = sph[w][j * 3 + 0];
      float h1 = sph[w][j * 3 + 1];
      float h2 = sph[w][j * 3 + 2];
      h4 hw;
      float4 vo;
      #pragma unroll
      for (int u = 0; u < 4; ++u) {
        int cc = c + u;
        float pr = spb2[cc] + h0 * sp2[cc * 3 + 0]
                            + h1 * sp2[cc * 3 + 1]
                            + h2 * sp2[cc * 3 + 2];
        float wp = (&kk.x)[u] - (&qq.x)[u] + pr;
        hw[u] = (_Float16)fmaxf(wp * sbn1s[cc] + sbn1b[cc], 0.0f);
        (&vo.x)[u] = (&vv.x)[u] + pr;
      }
      *(h4*)&swpre[w][j * CCH + c]     = hw;   // 8B packed-half store
      *(float4*)&svbuf[w][j * CCH + c] = vo;   // 16B store
    }
  }
  __syncthreads();

  const int row = lane & 15, hi = lane >> 4, col = row, hc = col & 7;

  // ---- GEMM1: (16 nbrs x 64) @ (64 x 8, zero-padded to 16 cols) ----
  v16h a0, a1;
  #pragma unroll
  for (int e = 0; e < 16; ++e) {
    int kl = 16 * hi + e;
    a0[e] = swpre[w][row * CCH + kl];
    a1[e] = swpre[w][row * CCH + 32 + kl];
  }
  const int kbb = 16 * hi;
  v16h b0 = {}, b1 = {};
  if (col < 8) {
    #pragma unroll
    for (int e = 0; e < 16; ++e) {
      b0[e] = sw1h[col * CCH + kbb + e];
      b1[e] = sw1h[col * CCH + 32 + kbb + e];
    }
  }
  v8f acc = {};
  acc = wmma_f16(a0, b0, acc);
  acc = wmma_f16(a1, b1, acc);

  float wb1 = w_b1[hc], s2 = sbn2s[hc], bb2 = sbn2b[hc];
  #pragma unroll
  for (int i = 0; i < 8; ++i) {
    float v1 = fmaxf((acc[i] + wb1) * s2 + bb2, 0.0f);
    if (col < 8) sw1buf[w][(i + 8 * hi) * 8 + col] = v1;
  }
  __syncthreads();

  // ---- GEMM2: (16 x 8, K zero-padded to 32) @ (8 x 8 padded) ----
  v16h a2 = {}, b2f = {};
  if (hi == 0) {
    #pragma unroll
    for (int e = 0; e < 8; ++e) a2[e] = (_Float16)sw1buf[w][row * 8 + e];  // K=e
    if (col < 8) {
      #pragma unroll
      for (int e = 0; e < 8; ++e) b2f[e] = sw2h[col * 8 + e];              // K=e
    }
  }
  v8f acc2 = {};
  acc2 = wmma_f16(a2, b2f, acc2);

  // ---- softmax over neighbors (M dim = my 8 rows + partner lane's 8) ----
  float wb2 = w_b2[hc];
  float lv[8];
  float mx = -1e30f;
  #pragma unroll
  for (int i = 0; i < 8; ++i) { lv[i] = acc2[i] + wb2; mx = fmaxf(mx, lv[i]); }
  mx = fmaxf(mx, __shfl_xor(mx, 16, 32));
  float sum = 0.0f;
  #pragma unroll
  for (int i = 0; i < 8; ++i) { lv[i] = __expf(lv[i] - mx); sum += lv[i]; }
  sum += __shfl_xor(sum, 16, 32);
  float inv = 1.0f / sum;
  #pragma unroll
  for (int i = 0; i < 8; ++i)
    if (col < 8) swn[w][(i + 8 * hi) * 8 + col] = lv[i] * inv;
  __syncthreads();

  // ---- grouped weighted reduce: out[c] = sum_j v[j][c] * w[j][c&7] ----
  #pragma unroll
  for (int half = 0; half < 2; ++half) {
    int c = lane + 32 * half;
    float o = 0.0f;
    #pragma unroll
    for (int j = 0; j < NSB; ++j)
      o += svbuf[w][j * CCH + c] * swn[w][j * 8 + (c & 7)];
    out[(size_t)n * CCH + c] = o;
  }
}

// ---------------------------------------------------------------------------
extern "C" void kernel_launch(void* const* d_in, const int* in_sizes, int n_in,
                              void* d_out, int out_size, void* d_ws, size_t ws_size,
                              hipStream_t stream) {
  const float*     p    = (const float*)d_in[0];
  const float*     x    = (const float*)d_in[1];
  const long long* idx  = (const long long*)d_in[2];   // jnp.int64
  const float* Wq = (const float*)d_in[3];
  const float* bq = (const float*)d_in[4];
  const float* Wk = (const float*)d_in[5];
  const float* bk = (const float*)d_in[6];
  const float* Wv = (const float*)d_in[7];
  const float* bv = (const float*)d_in[8];
  const float* p_w1   = (const float*)d_in[9];
  const float* p_b1   = (const float*)d_in[10];
  const float* p_bn_g = (const float*)d_in[11];
  const float* p_bn_b = (const float*)d_in[12];
  const float* p_bn_m = (const float*)d_in[13];
  const float* p_bn_v = (const float*)d_in[14];
  const float* p_w2   = (const float*)d_in[15];
  const float* p_b2   = (const float*)d_in[16];
  const float* w_bn1_g = (const float*)d_in[17];
  const float* w_bn1_b = (const float*)d_in[18];
  const float* w_bn1_m = (const float*)d_in[19];
  const float* w_bn1_v = (const float*)d_in[20];
  const float* w_w1    = (const float*)d_in[21];
  const float* w_b1    = (const float*)d_in[22];
  const float* w_bn2_g = (const float*)d_in[23];
  const float* w_bn2_b = (const float*)d_in[24];
  const float* w_bn2_m = (const float*)d_in[25];
  const float* w_bn2_v = (const float*)d_in[26];
  const float* w_w2    = (const float*)d_in[27];
  const float* w_b2    = (const float*)d_in[28];

  const int npts = in_sizes[1] / CCH;

  // workspace: xq | xk | xv  (3 * N * 64 floats; fits global L2 -> fast gather)
  float* xq = (float*)d_ws;
  float* xk = xq + (size_t)npts * CCH;
  float* xv = xk + (size_t)npts * CCH;

  int tiles = (npts + 15) / 16;
  int blkA  = (tiles + 7) / 8;
  proj_qkv_wmma<<<blkA, 256, 0, stream>>>(x, Wq, bq, Wk, bk, Wv, bv,
                                          xq, xk, xv, npts);

  int blkB = (npts + 7) / 8;
  attn_fused_wmma<<<blkB, 256, 0, stream>>>(
      p, idx, xq, xk, xv,
      p_w1, p_b1, p_bn_g, p_bn_b, p_bn_m, p_bn_v, p_w2, p_b2,
      w_bn1_g, w_bn1_b, w_bn1_m, w_bn1_v, w_w1, w_b1,
      w_bn2_g, w_bn2_b, w_bn2_m, w_bn2_v, w_w2, w_b2,
      (float*)d_out, npts);
}
